// GAT_80204219286053
// MI455X (gfx1250) — matrix operand order repaired and checked by reading, not compiled, not run on hardware
//
#include <hip/hip_runtime.h>
#include <hip/hip_bf16.h>
#include <stdint.h>

// ---------------------------------------------------------------------------
// GAT forward, 2 layers, N=8192, Fin=512, H1=256, H2=128 on gfx1250 (wave32).
//   0) pack adj (int32) -> 1-bit mask  (256MB -> 8MB, read once)
//   1) h1 = x @ W1                      (bf16 WMMA, f32 acc)
//   1b) pack h1 -> bf16 kpair tiles (LDS image layout) for async staging
//   2) src1/dst1 = h1 @ a1 halves
//   3) per-row online softmax stats (max, sum) over masked lrelu(src+dst^T)
//   4) acc1 += P1 @ h1  (P in-register; H tile staged to LDS via
//      global_load_async_to_lds_b128; K-split combined with f32 atomics)
//   5) o1 = relu(acc1)          [relu(elu(y)) == relu(y)]
//   6..10) same for layer 2; out = elu(acc2)
// ---------------------------------------------------------------------------

typedef __attribute__((ext_vector_type(16))) __bf16 v16bf;
typedef __attribute__((ext_vector_type(8)))  float  v8f;

#define NROWS   8192
#define MASKW   (NROWS / 32)
#define NEG_BIG (-9.0e15f)
#define ALPHA   0.2f
#define LDS_PAD 20   // words per column: 16 kpairs + pad; b128 stays 16B-aligned

__device__ __forceinline__ uint32_t pack_bf16(float lo, float hi) {
  union { uint32_t u; __bf16 h[2]; } c;
  c.h[0] = (__bf16)lo; c.h[1] = (__bf16)hi;
  return c.u;
}

// ---------------- 0) adj -> bitmask -----------------------------------------
__global__ __launch_bounds__(256) void k_pack_adj(const int* __restrict__ adj,
                                                  uint32_t* __restrict__ mask) {
  const int w = blockIdx.x * blockDim.x + threadIdx.x;   // one 32-bit word/thread
  const int4* p = reinterpret_cast<const int4*>(adj) + (size_t)w * 8;
  uint32_t m = 0;
#pragma unroll
  for (int q = 0; q < 8; ++q) {
    int4 v = p[q];
    const int b = q * 4;
    m |= (v.x > 0 ? 1u : 0u) << (b + 0);
    m |= (v.y > 0 ? 1u : 0u) << (b + 1);
    m |= (v.z > 0 ? 1u : 0u) << (b + 2);
    m |= (v.w > 0 ? 1u : 0u) << (b + 3);
  }
  mask[w] = m;
}

// ---------------- 1/6) C = A(MxK) @ B(KxN), one wave per 16x16 tile ----------
template <int K, int Nn>
__global__ __launch_bounds__(256) void k_gemm(const float* __restrict__ A,
                                              const float* __restrict__ B,
                                              float* __restrict__ C) {
  const int wid  = (blockIdx.x * blockDim.x + threadIdx.x) >> 5;
  const int lane = threadIdx.x & 31;
  const int mt   = wid / (Nn / 16);
  const int nt   = wid % (Nn / 16);
  const int half = lane >> 4;
  const int l15  = lane & 15;
  const int row  = mt * 16 + l15;
  const int n    = nt * 16 + l15;

  v8f acc = {};
  for (int k0 = 0; k0 < K; k0 += 32) {
    // A fragment: per-lane K runs {half*8..+7} and {16+half*8..+7} -> 4x b128
    union { float4 q[4]; float f[16]; } Af;
    const float* ap = &A[(size_t)row * K + k0 + half * 8];
    Af.q[0] = *(const float4*)(ap);
    Af.q[1] = *(const float4*)(ap + 4);
    Af.q[2] = *(const float4*)(ap + 16);
    Af.q[3] = *(const float4*)(ap + 20);
    v16bf af, bf;
#pragma unroll
    for (int r = 0; r < 8; ++r) {
      af[2 * r]     = (__bf16)Af.f[(r >> 2) * 8 + (r & 3) * 2];
      af[2 * r + 1] = (__bf16)Af.f[(r >> 2) * 8 + (r & 3) * 2 + 1];
      const int kb = k0 + (r >> 2) * 16 + half * 8 + (r & 3) * 2;   // B 32x16: K runs {half*8..+7},{16+half*8..+7}
      bf[2 * r]     = (__bf16)B[(size_t)kb * Nn + n];
      bf[2 * r + 1] = (__bf16)B[(size_t)(kb + 1) * Nn + n];
    }
    acc = __builtin_amdgcn_wmma_f32_16x16x32_bf16(false, af, false, bf,
                                                  (short)0, acc, false, false);
  }
#pragma unroll
  for (int r = 0; r < 8; ++r)
    C[(size_t)(mt * 16 + half * 8 + r) * Nn + n] = acc[r];
}

// ---------------- 1b/6b) h (f32) -> packed bf16 kpair tiles ------------------
// Tile jt holds K rows [32jt, 32jt+32). Word (jt, n, kp) = {bf16 h[32jt+2kp][n],
// bf16 h[32jt+2kp+1][n]} at hpk[jt*Fo*16 + n*16 + kp]  (the LDS tile image,
// minus the per-column pad which the async loads re-insert).
template <int Fo>
__global__ __launch_bounds__(256) void k_packh(const float* __restrict__ h,
                                               uint32_t* __restrict__ hpk) {
  const int tid = blockIdx.x * blockDim.x + threadIdx.x;   // (kq, n)
  const int kq  = tid / Fo;                                // kpair 0..4095
  const int n   = tid % Fo;
  const float lo = h[(size_t)(2 * kq) * Fo + n];
  const float hi = h[(size_t)(2 * kq + 1) * Fo + n];
  hpk[(size_t)(kq >> 4) * (Fo * 16) + n * 16 + (kq & 15)] = pack_bf16(lo, hi);
}

// ---------------- 2/7) src/dst = h @ a halves (one wave per row) -------------
template <int Fo>
__global__ __launch_bounds__(256) void k_srcdst(const float* __restrict__ h,
                                                const float* __restrict__ a,
                                                float* __restrict__ src,
                                                float* __restrict__ dst) {
  const int row  = (blockIdx.x * blockDim.x + threadIdx.x) >> 5;
  const int lane = threadIdx.x & 31;
  float s0 = 0.f, s1 = 0.f;
#pragma unroll
  for (int f = lane; f < Fo; f += 32) {
    const float hv = h[(size_t)row * Fo + f];
    s0 += hv * a[f];
    s1 += hv * a[Fo + f];
  }
#pragma unroll
  for (int off = 16; off; off >>= 1) {
    s0 += __shfl_xor(s0, off, 32);
    s1 += __shfl_xor(s1, off, 32);
  }
  if (lane == 0) { src[row] = s0; dst[row] = s1; }
}

// ---------------- 3/8) per-row online softmax stats --------------------------
__global__ __launch_bounds__(256) void k_stats(const float* __restrict__ src,
                                               const float* __restrict__ dst,
                                               const uint32_t* __restrict__ mask,
                                               float* __restrict__ rowm,
                                               float* __restrict__ rowinv) {
  const int row  = (blockIdx.x * blockDim.x + threadIdx.x) >> 5;
  const int lane = threadIdx.x & 31;
  const float sb = src[row];
  float m = NEG_BIG, s = 0.f;
  for (int it = 0; it < MASKW; ++it) {
    const uint32_t w = mask[(size_t)row * MASKW + it];
    float e = sb + dst[it * 32 + lane];
    e = e > 0.f ? e : ALPHA * e;
    e = ((w >> lane) & 1u) ? e : NEG_BIG;       // masked entries behave as -9e15
    const float nm = fmaxf(m, e);
    s = s * __expf(m - nm) + __expf(e - nm);    // exact all-masked semantics too
    m = nm;
  }
#pragma unroll
  for (int off = 16; off; off >>= 1) {
    const float om = __shfl_xor(m, off, 32);
    const float os = __shfl_xor(s, off, 32);
    const float nm = fmaxf(m, om);
    s = s * __expf(m - nm) + os * __expf(om - nm);
    m = nm;
  }
  if (lane == 0) { rowm[row] = m; rowinv[row] = 1.0f / s; }
}

// ---------------- 4/9) fused masked-softmax @ H (flash-style, K-split) -------
// Block = 8 waves sharing one K-chunk g, owning 8 consecutive M-tiles.
// Each K-step the block DMAs the pre-packed 32xFo bf16 tile into LDS with
// global_load_async_to_lds_b128 (ASYNCcnt), re-inserting the per-column pad
// via per-lane LDS addresses. B fragment = 2x ds_load_b128 bit-cast to v16bf.
// P fragment built in-register from src/dst/mask + exp. f32 atomics combine
// the G K-chunks (weights are pre-normalized by 1/sum).
template <int NT, int G>
__global__ __launch_bounds__(256) void k_att(const uint32_t* __restrict__ hpk,
                                             const float* __restrict__ src,
                                             const float* __restrict__ dst,
                                             const float* __restrict__ rowm,
                                             const float* __restrict__ rowinv,
                                             const uint32_t* __restrict__ mask,
                                             float* __restrict__ out) {
  constexpr int Fo = NT * 16;
  __shared__ uint32_t lds[Fo * LDS_PAD];

  const int t    = threadIdx.x;
  const int lane = t & 31;
  const int wv   = t >> 5;                       // wave in block: 0..7
  const int g    = blockIdx.x % G;               // K-chunk (shared by block)
  const int mt   = (blockIdx.x / G) * 8 + wv;    // M-tile per wave
  const int half = lane >> 4;
  const int l15  = lane & 15;
  const int row  = mt * 16 + l15;

  const float sb = src[row];
  const float mr = rowm[row];
  const float ir = rowinv[row];

  v8f acc[NT];
#pragma unroll
  for (int q = 0; q < NT; ++q) acc[q] = (v8f){};

  const int kspan = NROWS / G;
  const int kbeg  = g * kspan;
  for (int jb = kbeg; jb < kbeg + kspan; jb += 32) {
    // ---- async DMA: packed tile (Fo*16 words) -> LDS, pad re-inserted ----
    const uint32_t* gt = hpk + (size_t)(jb >> 5) * (Fo * 16);
#pragma unroll
    for (int c = 0; c < Fo / 64; ++c) {
      const int q  = t + 256 * c;                 // b128 chunk id, 0..Fo*4-1
      const int n  = q >> 2;                      // column
      const int cc = q & 3;                       // 4-word chunk in column
      const uint32_t* gsrc = gt + n * 16 + cc * 4;
      const uint32_t  ldst =
          (uint32_t)(uintptr_t)(&lds[n * LDS_PAD + cc * 4]);
      asm volatile("global_load_async_to_lds_b128 %0, %1, off"
                   :: "v"(ldst), "v"(gsrc) : "memory");
    }
    if (jb + 32 < kbeg + kspan)                   // cover next tile's latency
      __builtin_prefetch(gt + Fo * 16 + (size_t)t * (Fo * 16 / 256), 0, 0);

    // ---- P fragment while the DMA flies: dst runs contiguous -> 4x b128 ----
    union { float4 q[4]; float f[16]; } D;
    const float* dp = &dst[jb + half * 8];
    D.q[0] = *(const float4*)(dp);
    D.q[1] = *(const float4*)(dp + 4);
    D.q[2] = *(const float4*)(dp + 16);
    D.q[3] = *(const float4*)(dp + 20);
    const uint32_t mw = mask[(size_t)row * MASKW + (jb >> 5)];
    v16bf pf;
#pragma unroll
    for (int r = 0; r < 8; ++r) {
#pragma unroll
      for (int u = 0; u < 2; ++u) {
        const int fi = (r >> 2) * 8 + (r & 3) * 2 + u;              // 0..15
        const int ko = (r >> 2) * 16 + (r & 3) * 2 + half * 8 + u;  // 0..31
        float e = sb + D.f[fi];
        e = e > 0.f ? e : ALPHA * e;
        e = ((mw >> ko) & 1u) ? e : NEG_BIG;
        pf[2 * r + u] = (__bf16)(__expf(e - mr) * ir);
      }
    }

    asm volatile("s_wait_asynccnt 0x0" ::: "memory");
    __syncthreads();

    // ---- NT WMMAs; B fragment = 2x ds_load_b128 bit-cast to v16bf ----
#pragma unroll
    for (int nt = 0; nt < NT; ++nt) {
      const uint32_t* lp = &lds[(size_t)(nt * 16 + l15) * LDS_PAD + half * 4];
      union { uint4 q[2]; v16bf v; } H;
      H.q[0] = *(const uint4*)(lp);
      H.q[1] = *(const uint4*)(lp + 8);
      acc[nt] = __builtin_amdgcn_wmma_f32_16x16x32_bf16(false, pf, false, H.v,
                                                        (short)0, acc[nt],
                                                        false, false);
    }
    __syncthreads();   // protect LDS from next iteration's DMA
  }
#pragma unroll
  for (int nt = 0; nt < NT; ++nt)
#pragma unroll
    for (int r = 0; r < 8; ++r)
      atomicAdd(&out[(size_t)(mt * 16 + half * 8 + r) * Fo + nt * 16 + l15],
                acc[nt][r]);
}

// ---------------- 5/10) activations ------------------------------------------
__global__ __launch_bounds__(256) void k_relu(const float* __restrict__ in,
                                              float* __restrict__ out) {
  const int i = blockIdx.x * blockDim.x + threadIdx.x;
  out[i] = fmaxf(in[i], 0.f);
}
__global__ __launch_bounds__(256) void k_elu(const float* __restrict__ in,
                                             float* __restrict__ out) {
  const int i = blockIdx.x * blockDim.x + threadIdx.x;
  const float v = in[i];
  out[i] = v > 0.f ? v : (__expf(v) - 1.f);
}

// ---------------------------------------------------------------------------
extern "C" void kernel_launch(void* const* d_in, const int* in_sizes, int n_in,
                              void* d_out, int out_size, void* d_ws, size_t ws_size,
                              hipStream_t stream) {
  const float* x  = (const float*)d_in[0];
  const int*   adj= (const int*)  d_in[1];
  const float* W1 = (const float*)d_in[2];
  const float* a1 = (const float*)d_in[3];
  const float* W2 = (const float*)d_in[4];
  const float* a2 = (const float*)d_in[5];
  float* out = (float*)d_out;

  char* ws = (char*)d_ws;
  const size_t MB = 1024u * 1024u;
  uint32_t* mask = (uint32_t*)(ws + 0 * MB);   // 8 MB
  float* h1   = (float*)(ws +  8 * MB);        // 8 MB  (8192x256)
  float* acc1 = (float*)(ws + 16 * MB);        // 8 MB
  float* o1   = (float*)(ws + 24 * MB);        // 8 MB
  float* h2   = (float*)(ws + 32 * MB);        // 4 MB  (8192x128)
  float* acc2 = (float*)(ws + 36 * MB);        // 4 MB
  float* vec  = (float*)(ws + 40 * MB);        // stats vectors (256 KB)
  uint32_t* hpk1 = (uint32_t*)(ws + 41 * MB);  // 4 MB packed bf16 tiles
  uint32_t* hpk2 = (uint32_t*)(ws + 45 * MB);  // 2 MB packed bf16 tiles
  float *src1 = vec,             *dst1 = vec + 8192,
        *m1   = vec + 2 * 8192,  *i1   = vec + 3 * 8192,
        *src2 = vec + 4 * 8192,  *dst2 = vec + 5 * 8192,
        *m2   = vec + 6 * 8192,  *i2   = vec + 7 * 8192;

  // 0) pack adjacency: 2,097,152 words, one per thread
  k_pack_adj<<<(NROWS * MASKW) / 256, 256, 0, stream>>>(adj, mask);

  // ---- layer 1 ----
  k_gemm<512, 256><<<(512 * 16) / 8, 256, 0, stream>>>(x, W1, h1);
  k_packh<256><<<(NROWS / 2) * 256 / 256, 256, 0, stream>>>(h1, hpk1);
  k_srcdst<256><<<NROWS / 8, 256, 0, stream>>>(h1, a1, src1, dst1);
  k_stats<<<NROWS / 8, 256, 0, stream>>>(src1, dst1, mask, m1, i1);
  hipMemsetAsync(acc1, 0, (size_t)NROWS * 256 * sizeof(float), stream);
  k_att<16, 8><<<8 * (512 / 8), 256, 0, stream>>>(hpk1, src1, dst1, m1, i1, mask, acc1);
  k_relu<<<(NROWS * 256) / 256, 256, 0, stream>>>(acc1, o1);

  // ---- layer 2 ----
  k_gemm<256, 128><<<(512 * 8) / 8, 256, 0, stream>>>(o1, W2, h2);
  k_packh<128><<<(NROWS / 2) * 128 / 256, 256, 0, stream>>>(h2, hpk2);
  k_srcdst<128><<<NROWS / 8, 256, 0, stream>>>(h2, a2, src2, dst2);
  k_stats<<<NROWS / 8, 256, 0, stream>>>(src2, dst2, mask, m2, i2);
  hipMemsetAsync(acc2, 0, (size_t)NROWS * 128 * sizeof(float), stream);
  k_att<8, 8><<<8 * (512 / 8), 256, 0, stream>>>(hpk2, src2, dst2, m2, i2, mask, acc2);
  k_elu<<<(NROWS * 128) / 256, 256, 0, stream>>>(acc2, out);
}
